// MMSegSwinTransformer_7404523618526
// MI455X (gfx1250) — compile-verified, matrix operand order checked by reading, not run on hardware
//
#include <hip/hip_runtime.h>
#include <hip/hip_bf16.h>
#include <math.h>

typedef __attribute__((ext_vector_type(8)))  _Float16 v8h;
typedef __attribute__((ext_vector_type(16))) _Float16 v16h;
typedef __attribute__((ext_vector_type(8)))  float    v8f;
typedef __attribute__((ext_vector_type(4)))  int      v4i;

#if __has_builtin(__builtin_amdgcn_global_load_async_to_lds_b128) && \
    __has_builtin(__builtin_amdgcn_s_wait_asynccnt)
#define USE_ASYNC_LDS 1
#else
#define USE_ASYNC_LDS 0
#endif

// ---------------------------------------------------------------------------
// helpers
// ---------------------------------------------------------------------------
__device__ __forceinline__ float gelu_exact(float x) {
  return 0.5f * x * (1.0f + erff(x * 0.70710678118654752f));
}

#if USE_ASYNC_LDS
// builtin expects: (v4i __device__* /*AS1*/, v4i __shared__* /*AS3*/, imm, imm)
__device__ __forceinline__ __attribute__((address_space(1))) v4i* to_glob(const void* p) {
  return (__attribute__((address_space(1))) v4i*)(uintptr_t)p;
}
__device__ __forceinline__ __attribute__((address_space(3))) v4i* to_lds(void* p) {
  return (__attribute__((address_space(3))) v4i*)(unsigned)(uintptr_t)p;
}
#endif

// Load a 16x32 f16 fragment (A-layout, per CDNA5 ISA 7.12.2) from LDS.
// base = row 0 / k 0 of the tile, ld = row pitch in halves (multiple of 8).
// lanes 0-15: row = lane,   k in [0..7] ++ [16..23]
// lanes 16-31: row = lane-16, k in [8..15] ++ [24..31]
__device__ __forceinline__ v16h ldsfrag(const _Float16* base, int ld) {
  int lane = threadIdx.x & 31;
  int row  = lane & 15;
  int ksel = (lane >> 4) << 3;
  const _Float16* p = base + row * ld + ksel;
  v8h lo = *(const v8h*)(p);
  v8h hi = *(const v8h*)(p + 16);
  return __builtin_shufflevector(lo, hi, 0, 1, 2, 3, 4, 5, 6, 7,
                                 8, 9, 10, 11, 12, 13, 14, 15);
}

// ---------------------------------------------------------------------------
// f32 -> f16 weight conversion
// ---------------------------------------------------------------------------
__global__ void f32_to_f16_k(const float* __restrict__ in,
                             _Float16* __restrict__ out, long long n) {
  long long i = (long long)blockIdx.x * blockDim.x + threadIdx.x;
  if (i < n) out[i] = (_Float16)in[i];
}

// ---------------------------------------------------------------------------
// Generic WMMA GEMM:  C[M,N] = act(A[M,K] * W[N,K]^T + bias) (+res)
// A: f16 row-major, W: f16 row-major (out_features x in_features).
// 128 threads = 4 waves; 64x64 output tile; K stepped by 32.
// Interior tiles stage global->LDS with the CDNA5 async path (ASYNCcnt);
// boundary tiles fall back to the VGPR path with zero padding.
// ---------------------------------------------------------------------------
template <int ACT, typename OutT, bool RESID>
__global__ __launch_bounds__(128) void gemm_wmma(
    const _Float16* __restrict__ A, const _Float16* __restrict__ W,
    const float* __restrict__ bias, const float* __restrict__ res,
    OutT* __restrict__ C, int M, int N, int K) {
  __shared__ _Float16 As[64][40];
  __shared__ _Float16 Ws[64][40];
  const int tid = threadIdx.x;
  const int wv = tid >> 5, lane = tid & 31;
  const int m0 = blockIdx.y * 64, n0 = blockIdx.x * 64;
  v8f acc[4] = {};

  const bool interior = (m0 + 64 <= M) && (n0 + 64 <= N);

  for (int k0 = 0; k0 < K; k0 += 32) {
#if USE_ASYNC_LDS
    if (interior) {
      // async DMA: 2 x b128 per thread per tile, no VGPR round-trip
#pragma unroll
      for (int it = 0; it < 2; ++it) {
        int c = tid + it * 128;
        int r = c >> 2, kk = (c & 3) << 3;
        __builtin_amdgcn_global_load_async_to_lds_b128(
            to_glob(A + (size_t)(m0 + r) * K + k0 + kk), to_lds(&As[r][kk]), 0, 0);
        __builtin_amdgcn_global_load_async_to_lds_b128(
            to_glob(W + (size_t)(n0 + r) * K + k0 + kk), to_lds(&Ws[r][kk]), 0, 0);
      }
      __builtin_amdgcn_s_wait_asynccnt(0);
    } else
#endif
    {
      // stage A tile (64 rows x 32 halves), zero-pad OOB rows
      for (int c = tid; c < 256; c += 128) {
        int r = c >> 2, kk = (c & 3) << 3;
        v8h val = {};
        int gm = m0 + r;
        if (gm < M) {
          val = *(const v8h*)(A + (size_t)gm * K + k0 + kk);
          if (k0 + 32 < K) __builtin_prefetch(A + (size_t)gm * K + k0 + 32 + kk, 0, 0);
        }
        *(v8h*)(&As[r][kk]) = val;
      }
      // stage W tile (64 rows x 32 halves)
      for (int c = tid; c < 256; c += 128) {
        int r = c >> 2, kk = (c & 3) << 3;
        v8h val = {};
        int gn = n0 + r;
        if (gn < N) {
          val = *(const v8h*)(W + (size_t)gn * K + k0 + kk);
          if (k0 + 32 < K) __builtin_prefetch(W + (size_t)gn * K + k0 + 32 + kk, 0, 0);
        }
        *(v8h*)(&Ws[r][kk]) = val;
      }
    }
    __syncthreads();
    v16h a = ldsfrag(&As[wv * 16][0], 40);
#pragma unroll
    for (int t = 0; t < 4; ++t) {
      v16h b = ldsfrag(&Ws[t * 16][0], 40);
      acc[t] = __builtin_amdgcn_wmma_f32_16x16x32_f16(
          false, a, false, b, (short)0, acc[t], false, false);
    }
    __syncthreads();
  }

  // epilogue: bias + activation + optional residual, per ISA C/D layout
  const int mbase = m0 + wv * 16 + ((lane >> 4) << 3);
  const int nl = lane & 15;
#pragma unroll
  for (int t = 0; t < 4; ++t) {
    int n = n0 + t * 16 + nl;
    if (n >= N) continue;
    float bv = bias ? bias[n] : 0.0f;
#pragma unroll
    for (int r = 0; r < 8; ++r) {
      int m = mbase + r;
      if (m >= M) continue;
      float v = acc[t][r] + bv;
      if (ACT == 1) v = gelu_exact(v);
      if (RESID) v += res[(size_t)m * N + n];
      C[(size_t)m * N + n] = (OutT)v;
    }
  }
}

// ---------------------------------------------------------------------------
// Fused windowed attention, one block per (window, head). N=49 (pad 64), hd=32.
// S = (q*scale) K^T + relpos(+mask); softmax; O = P V. All WMMA in LDS.
// ---------------------------------------------------------------------------
__global__ __launch_bounds__(128) void attn_wmma(
    const _Float16* __restrict__ qkv,  // (Bn, 49, 3C)
    const float* __restrict__ rpb,     // (169, nH)
    _Float16* __restrict__ out,        // (Bn, 49, C)
    int nH, int C, int nWside, int shifted, int Hres) {
  __shared__ _Float16 Qs[64][40];
  __shared__ _Float16 Ks[64][40];
  __shared__ _Float16 Vt[32][72];  // V transposed: [d][m]
  __shared__ _Float16 Ps[64][72];  // softmax probs (padded K=64)
  const int win = blockIdx.x;
  const int h = blockIdx.y;
  const int tid = threadIdx.x;
  const int wv = tid >> 5, lane = tid & 31;
  const int C3 = 3 * C;

  for (int idx = tid; idx < 64 * 32; idx += 128) {
    int m = idx >> 5, d = idx & 31;
    _Float16 qv = (_Float16)0.0f, kv = (_Float16)0.0f, vv = (_Float16)0.0f;
    if (m < 49) {
      const _Float16* base = qkv + ((size_t)win * 49 + m) * C3;
      qv = (_Float16)((float)base[h * 32 + d] * 0.17677669529663689f);  // hd^-0.5
      kv = base[(nH + h) * 32 + d];
      vv = base[(2 * nH + h) * 32 + d];
    }
    Qs[m][d] = qv;
    Ks[m][d] = kv;
    Vt[d][m] = vv;
  }
  __syncthreads();

  // S tile: rows [16*wv .. +15] x cols 0..63, single K=32 step
  v8f s[4] = {};
  {
    v16h a = ldsfrag(&Qs[wv * 16][0], 40);
#pragma unroll
    for (int t = 0; t < 4; ++t) {
      v16h b = ldsfrag(&Ks[t * 16][0], 40);
      s[t] = __builtin_amdgcn_wmma_f32_16x16x32_f16(
          false, a, false, b, (short)0, s[t], false, false);
    }
  }

  // relative position bias + shift mask, computed arithmetically
  const int nW = nWside * nWside;
  const int wq = win % nW;
  const int wy = wq / nWside, wx = wq % nWside;
  const int mbase = wv * 16 + ((lane >> 4) << 3);
  const int nl = lane & 15;
#pragma unroll
  for (int t = 0; t < 4; ++t) {
    int n = t * 16 + nl;
#pragma unroll
    for (int r = 0; r < 8; ++r) {
      int m = mbase + r;
      float v = s[t][r];
      if (m < 49 && n < 49) {
        int yi = m / 7, xi = m % 7, yj = n / 7, xj = n % 7;
        int ridx = (yi - yj + 6) * 13 + (xi - xj + 6);
        v += rpb[ridx * nH + h];
        if (shifted) {
          int gy = wy * 7 + yi, gx = wx * 7 + xi;
          int hy = wy * 7 + yj, hx = wx * 7 + xj;
          int rq = (gy < Hres - 7 ? 0 : (gy < Hres - 3 ? 1 : 2)) * 3 +
                   (gx < Hres - 7 ? 0 : (gx < Hres - 3 ? 1 : 2));
          int rk = (hy < Hres - 7 ? 0 : (hy < Hres - 3 ? 1 : 2)) * 3 +
                   (hx < Hres - 7 ? 0 : (hx < Hres - 3 ? 1 : 2));
          if (rq != rk) v += -100.0f;
        }
      } else {
        v = -1e30f;  // padding -> softmax weight 0
      }
      s[t][r] = v;
    }
  }

  // in-register softmax across the 16-lane row group (4 values per lane)
#pragma unroll
  for (int r = 0; r < 8; ++r) {
    float mx = s[0][r];
#pragma unroll
    for (int t = 1; t < 4; ++t) mx = fmaxf(mx, s[t][r]);
#pragma unroll
    for (int off = 1; off < 16; off <<= 1) mx = fmaxf(mx, __shfl_xor(mx, off, 32));
    float e[4], sum = 0.0f;
#pragma unroll
    for (int t = 0; t < 4; ++t) { e[t] = __expf(s[t][r] - mx); sum += e[t]; }
#pragma unroll
    for (int off = 1; off < 16; off <<= 1) sum += __shfl_xor(sum, off, 32);
    float inv = 1.0f / sum;
    int m = mbase + r;
#pragma unroll
    for (int t = 0; t < 4; ++t) Ps[m][t * 16 + nl] = (_Float16)(e[t] * inv);
  }
  __syncthreads();

  // O = P(64x64) * V(64x32): 2 K-steps, 2 N-subtiles
  v8f o[2] = {};
  {
    v16h a0 = ldsfrag(&Ps[wv * 16][0], 72);
    v16h a1 = ldsfrag(&Ps[wv * 16][32], 72);
#pragma unroll
    for (int t = 0; t < 2; ++t) {
      v16h b0 = ldsfrag(&Vt[t * 16][0], 72);
      v16h b1 = ldsfrag(&Vt[t * 16][32], 72);
      o[t] = __builtin_amdgcn_wmma_f32_16x16x32_f16(
          false, a0, false, b0, (short)0, o[t], false, false);
      o[t] = __builtin_amdgcn_wmma_f32_16x16x32_f16(
          false, a1, false, b1, (short)0, o[t], false, false);
    }
  }
#pragma unroll
  for (int t = 0; t < 2; ++t) {
    int d = t * 16 + nl;
#pragma unroll
    for (int r = 0; r < 8; ++r) {
      int m = mbase + r;
      if (m < 49) out[((size_t)win * 49 + m) * C + h * 32 + d] = (_Float16)o[t][r];
    }
  }
}

// ---------------------------------------------------------------------------
// LayerNorm over last dim C: one wave per row (4 rows / 128-thread block)
// ---------------------------------------------------------------------------
template <typename OutT>
__global__ __launch_bounds__(128) void layernorm_k(
    const float* __restrict__ in, const float* __restrict__ g,
    const float* __restrict__ b, OutT* __restrict__ out, int rows, int C) {
  int row = blockIdx.x * 4 + (threadIdx.x >> 5);
  if (row >= rows) return;
  int lane = threadIdx.x & 31;
  const float* p = in + (size_t)row * C;
  float s = 0.0f, s2 = 0.0f;
  for (int c = lane; c < C; c += 32) { float v = p[c]; s += v; s2 += v * v; }
#pragma unroll
  for (int off = 1; off < 32; off <<= 1) {
    s += __shfl_xor(s, off, 32);
    s2 += __shfl_xor(s2, off, 32);
  }
  float mean = s / (float)C;
  float var = s2 / (float)C - mean * mean;
  float inv = rsqrtf(var + 1e-5f);
  OutT* q = out + (size_t)row * C;
  for (int c = lane; c < C; c += 32)
    q[c] = (OutT)((p[c] - mean) * inv * g[c] + b[c]);
}

// ---------------------------------------------------------------------------
// patch embed: conv 4x4 stride 4, NCHW input -> (B, 56*56, 96) f32
// ---------------------------------------------------------------------------
__global__ void patch_embed_k(const float* __restrict__ x,
                              const float* __restrict__ w,
                              const float* __restrict__ bias,
                              float* __restrict__ out) {
  size_t total = (size_t)32 * 3136 * 96;
  size_t i = (size_t)blockIdx.x * blockDim.x + threadIdx.x;
  if (i >= total) return;
  int co = (int)(i % 96);
  size_t rest = i / 96;
  int ox = (int)(rest % 56); rest /= 56;
  int oy = (int)(rest % 56);
  int b = (int)(rest / 56);
  float acc = bias[co];
#pragma unroll
  for (int ci = 0; ci < 3; ++ci)
#pragma unroll
    for (int ky = 0; ky < 4; ++ky)
#pragma unroll
      for (int kx = 0; kx < 4; ++kx)
        acc += w[((co * 3 + ci) * 4 + ky) * 4 + kx] *
               x[(((size_t)b * 3 + ci) * 224 + (oy * 4 + ky)) * 224 + (ox * 4 + kx)];
  out[i] = acc;
}

// ---------------------------------------------------------------------------
// window partition with cyclic roll(-shift): (B,H,W,C) f32 -> (Bn,49,C) f16
// ---------------------------------------------------------------------------
__global__ void win_partition_k(const float* __restrict__ in,
                                _Float16* __restrict__ out, int B, int H, int C,
                                int shift) {
  size_t total = (size_t)B * H * H * C;
  size_t i = (size_t)blockIdx.x * blockDim.x + threadIdx.x;
  if (i >= total) return;
  int c = (int)(i % C);
  size_t rest = i / C;
  int xr = (int)(rest % H); rest /= H;
  int yr = (int)(rest % H);
  int b = (int)(rest / H);
  int ys = (yr + shift) % H, xs = (xr + shift) % H;
  int nWs = H / 7;
  int winidx = (b * nWs + yr / 7) * nWs + xr / 7;
  int pix = (yr % 7) * 7 + (xr % 7);
  out[((size_t)winidx * 49 + pix) * C + c] =
      (_Float16)in[(((size_t)b * H + ys) * H + xs) * C + c];
}

// window reverse + roll(+shift) + residual add into X
__global__ void win_reverse_add_k(const float* __restrict__ winb,
                                  float* __restrict__ X, int B, int H, int C,
                                  int shift) {
  size_t total = (size_t)B * H * H * C;
  size_t i = (size_t)blockIdx.x * blockDim.x + threadIdx.x;
  if (i >= total) return;
  int c = (int)(i % C);
  size_t rest = i / C;
  int xr = (int)(rest % H); rest /= H;
  int yr = (int)(rest % H);
  int b = (int)(rest / H);
  int ys = (yr + shift) % H, xs = (xr + shift) % H;
  int nWs = H / 7;
  int winidx = (b * nWs + yr / 7) * nWs + xr / 7;
  int pix = (yr % 7) * 7 + (xr % 7);
  X[(((size_t)b * H + ys) * H + xs) * C + c] +=
      winb[((size_t)winidx * 49 + pix) * C + c];
}

// patch merging rearrange: (B,H,W,C) -> (B, H/2*W/2, 4C), ch = c*4 + i*2 + j
__global__ void merge_rearrange_k(const float* __restrict__ X,
                                  float* __restrict__ out, int B, int H, int C) {
  int C4 = 4 * C, H2 = H / 2;
  size_t total = (size_t)B * H2 * H2 * C4;
  size_t i = (size_t)blockIdx.x * blockDim.x + threadIdx.x;
  if (i >= total) return;
  int cc = (int)(i % C4);
  size_t rest = i / C4;
  int x2 = (int)(rest % H2); rest /= H2;
  int y2 = (int)(rest % H2);
  int b = (int)(rest / H2);
  int c = cc >> 2, ii = (cc >> 1) & 1, jj = cc & 1;
  out[i] = X[(((size_t)b * H + (2 * y2 + ii)) * H + (2 * x2 + jj)) * C + c];
}

// final LN + (B,49,768) -> (B,768,7,7) transpose
__global__ __launch_bounds__(128) void final_ln_transpose_k(
    const float* __restrict__ in, const float* __restrict__ g,
    const float* __restrict__ b, float* __restrict__ out) {
  int row = blockIdx.x * 4 + (threadIdx.x >> 5);
  if (row >= 32 * 49) return;
  int lane = threadIdx.x & 31;
  const float* p = in + (size_t)row * 768;
  float s = 0.0f, s2 = 0.0f;
  for (int c = lane; c < 768; c += 32) { float v = p[c]; s += v; s2 += v * v; }
#pragma unroll
  for (int off = 1; off < 32; off <<= 1) {
    s += __shfl_xor(s, off, 32);
    s2 += __shfl_xor(s2, off, 32);
  }
  float mean = s / 768.0f;
  float inv = rsqrtf(s2 / 768.0f - mean * mean + 1e-5f);
  int bi = row / 49, pix = row % 49, hh = pix / 7, ww = pix % 7;
  for (int c = lane; c < 768; c += 32) {
    float v = (p[c] - mean) * inv * g[c] + b[c];
    out[(((size_t)bi * 768 + c) * 7 + hh) * 7 + ww] = v;
  }
}

// ---------------------------------------------------------------------------
// host orchestration
// ---------------------------------------------------------------------------
extern "C" void kernel_launch(void* const* d_in, const int* in_sizes, int n_in,
                              void* d_out, int out_size, void* d_ws,
                              size_t ws_size, hipStream_t stream) {
  (void)in_sizes; (void)n_in; (void)out_size; (void)ws_size;
  static const int DIMS_[4] = {96, 192, 384, 768};
  static const int HEADS_[4] = {3, 6, 12, 24};
  static const int DEPTHS_[4] = {2, 2, 6, 2};
  static const int RES_[4] = {56, 28, 14, 7};
  const int B = 32;

  // ---- input index map (jax tree_flatten: dict keys sorted, lists in order)
  int idx = 0;
  const int i_out_ln_b = idx++;
  const int i_out_ln_g = idx++;
  const int i_pe_b = idx++;
  const int i_pe_ln_b = idx++;
  const int i_pe_ln_g = idx++;
  const int i_pe_w = idx++;
  int blk[4][6][13];
  int mrg[3][3];
  for (int s = 0; s < 4; ++s) {
    for (int d = 0; d < DEPTHS_[s]; ++d)
      for (int t = 0; t < 13; ++t) blk[s][d][t] = idx++;
    if (s < 3)
      for (int t = 0; t < 3; ++t) mrg[s][t] = idx++;
  }
  const int i_x = idx++;
  enum { T_FC1_B = 0, T_FC1_W, T_FC2_B, T_FC2_W, T_LN1_B, T_LN1_G, T_LN2_B,
         T_LN2_G, T_PROJ_B, T_PROJ_W, T_QKV_B, T_QKV_W, T_RPB };
  auto F = [&](int i) { return (const float*)d_in[i]; };

  // ---- workspace carve (256B aligned slabs)
  char* wp = (char*)d_ws;
  auto alloc = [&](size_t bytes) -> void* {
    void* p = (void*)wp;
    wp += (bytes + 255) & ~(size_t)255;
    return p;
  };
  const size_t maxE = (size_t)B * 3136 * 96;  // largest B*L*C
  float* X = (float*)alloc(maxE * 4);         // residual stream f32
  float* T = (float*)alloc(maxE * 4);         // f32 temp
  _Float16* AH = (_Float16*)alloc(maxE * 2);  // f16 GEMM input
  _Float16* QKV = (_Float16*)alloc(maxE * 3 * 2);
  _Float16* AO = (_Float16*)alloc(maxE * 2);  // attention output (windowed)
  _Float16* HID = (_Float16*)alloc(maxE * 4 * 2);  // MLP hidden

  // f16 weight cache
  auto cvt = [&](const float* src, size_t n) -> _Float16* {
    _Float16* dst = (_Float16*)alloc(n * 2);
    int grid = (int)((n + 255) / 256);
    f32_to_f16_k<<<grid, 256, 0, stream>>>(src, dst, (long long)n);
    return dst;
  };
  _Float16 *qkvW[4][6], *projW[4][6], *fc1W[4][6], *fc2W[4][6], *redW[3];
  for (int s = 0; s < 4; ++s) {
    int C = DIMS_[s];
    for (int d = 0; d < DEPTHS_[s]; ++d) {
      qkvW[s][d] = cvt(F(blk[s][d][T_QKV_W]), (size_t)3 * C * C);
      projW[s][d] = cvt(F(blk[s][d][T_PROJ_W]), (size_t)C * C);
      fc1W[s][d] = cvt(F(blk[s][d][T_FC1_W]), (size_t)4 * C * C);
      fc2W[s][d] = cvt(F(blk[s][d][T_FC2_W]), (size_t)4 * C * C);
    }
    if (s < 3) redW[s] = cvt(F(mrg[s][2]), (size_t)DIMS_[s + 1] * 4 * C);
  }

  // ---- patch embed + LN
  {
    size_t total = (size_t)B * 3136 * 96;
    patch_embed_k<<<(int)((total + 255) / 256), 256, 0, stream>>>(
        F(i_x), F(i_pe_w), F(i_pe_b), T);
    int rows = B * 3136;
    layernorm_k<float><<<(rows + 3) / 4, 128, 0, stream>>>(
        T, F(i_pe_ln_g), F(i_pe_ln_b), X, rows, 96);
  }

  // ---- stages
  for (int s = 0; s < 4; ++s) {
    const int H = RES_[s], C = DIMS_[s], nH = HEADS_[s];
    const int L = H * H, nWs = H / 7, nW = nWs * nWs;
    const int Mfull = B * L, Mwin = B * nW * 49;
    const size_t totalBHWC = (size_t)B * L * C;
    const int ebGrid = (int)((totalBHWC + 255) / 256);

    for (int d = 0; d < DEPTHS_[s]; ++d) {
      const int* bp = blk[s][d];
      const int shift = (d % 2) ? 3 : 0;

      // LN1 -> T (f32), then roll+window-partition -> AH (f16)
      layernorm_k<float><<<(Mfull + 3) / 4, 128, 0, stream>>>(
          X, F(bp[T_LN1_G]), F(bp[T_LN1_B]), T, Mfull, C);
      win_partition_k<<<ebGrid, 256, 0, stream>>>(T, AH, B, H, C, shift);

      // qkv GEMM (windowed): (Mwin x C) x (3C x C)^T
      {
        dim3 g((3 * C + 63) / 64, (Mwin + 63) / 64);
        gemm_wmma<0, _Float16, false><<<g, 128, 0, stream>>>(
            AH, qkvW[s][d], F(bp[T_QKV_B]), nullptr, QKV, Mwin, 3 * C, C);
      }

      // fused attention per (window, head)
      attn_wmma<<<dim3(B * nW, nH), 128, 0, stream>>>(
          QKV, F(bp[T_RPB]), AO, nH, C, nWs, shift ? 1 : 0, H);

      // proj GEMM -> T (f32, windowed), then reverse-roll + residual add -> X
      {
        dim3 g((C + 63) / 64, (Mwin + 63) / 64);
        gemm_wmma<0, float, false><<<g, 128, 0, stream>>>(
            AO, projW[s][d], F(bp[T_PROJ_B]), nullptr, T, Mwin, C, C);
      }
      win_reverse_add_k<<<ebGrid, 256, 0, stream>>>(T, X, B, H, C, shift);

      // MLP: LN2 -> AH (f16), fc1+GELU -> HID (f16), fc2+residual -> X (f32)
      layernorm_k<_Float16><<<(Mfull + 3) / 4, 128, 0, stream>>>(
          X, F(bp[T_LN2_G]), F(bp[T_LN2_B]), AH, Mfull, C);
      {
        dim3 g((4 * C + 63) / 64, (Mfull + 63) / 64);
        gemm_wmma<1, _Float16, false><<<g, 128, 0, stream>>>(
            AH, fc1W[s][d], F(bp[T_FC1_B]), nullptr, HID, Mfull, 4 * C, C);
      }
      {
        dim3 g((C + 63) / 64, (Mfull + 63) / 64);
        gemm_wmma<0, float, true><<<g, 128, 0, stream>>>(
            HID, fc2W[s][d], F(bp[T_FC2_B]), X, X, Mfull, C, 4 * C);
      }
    }

    if (s < 3) {
      const int C2 = DIMS_[s + 1];
      size_t totalM = (size_t)B * (L / 4) * 4 * C;
      merge_rearrange_k<<<(int)((totalM + 255) / 256), 256, 0, stream>>>(
          X, T, B, H, C);
      int rowsM = B * (L / 4);
      layernorm_k<_Float16><<<(rowsM + 3) / 4, 128, 0, stream>>>(
          T, F(mrg[s][1]), F(mrg[s][0]), AH, rowsM, 4 * C);
      dim3 g((C2 + 63) / 64, (rowsM + 63) / 64);
      gemm_wmma<0, float, false><<<g, 128, 0, stream>>>(
          AH, redW[s], nullptr, nullptr, X, rowsM, C2, 4 * C);
    }
  }

  // ---- final LN + transpose to (B,768,7,7)
  final_ln_transpose_k<<<(B * 49 + 3) / 4, 128, 0, stream>>>(
      X, F(i_out_ln_g), F(i_out_ln_b), (float*)d_out);
}
